// BiLSTM_CRF_4922032521710
// MI455X (gfx1250) — compile-verified, matrix operand order checked by reading, not run on hardware
//
#include <hip/hip_runtime.h>
#include <hip/hip_bf16.h>

typedef __bf16 bf16_t;
typedef __attribute__((ext_vector_type(16))) __bf16 v16bf;
typedef __attribute__((ext_vector_type(8)))  __bf16 v8bf;
typedef __attribute__((ext_vector_type(8)))  float  v8f;

namespace cfg {
constexpr int B = 32, T = 512, E = 512, H = 512, NK = 16;
constexpr int D = 513, DP = 544;          // D padded to multiple of 32 (WMMA K)
constexpr int G = 4 * H;                  // 2048 gate columns
constexpr int BT = B * T;                 // 16384
constexpr int H2 = 2 * H;                 // 1024

constexpr size_t al(size_t x) { return (x + 255) & ~(size_t)255; }
constexpr size_t SZ_XB   = (size_t)BT * DP * 2;
constexpr size_t SZ_WIHB = (size_t)G * DP * 2;
constexpr size_t SZ_WHHB = (size_t)G * H * 2;
constexpr size_t SZ_WOUT = (size_t)NK * H2 * 2;
constexpr size_t SZ_PRE  = (size_t)BT * G * 4;
constexpr size_t SZ_HBUF = (size_t)2 * 2 * B * H * 2;   // dir x pingpong x B x H
constexpr size_t SZ_LOUT = (size_t)BT * H2 * 2;
constexpr size_t SZ_FEAT = (size_t)BT * NK * 4;

constexpr size_t OFF_XB     = 0;
constexpr size_t OFF_WIHB_F = al(OFF_XB + SZ_XB);
constexpr size_t OFF_WIHB_B = al(OFF_WIHB_F + SZ_WIHB);
constexpr size_t OFF_WHHB_F = al(OFF_WIHB_B + SZ_WIHB);
constexpr size_t OFF_WHHB_B = al(OFF_WHHB_F + SZ_WHHB);
constexpr size_t OFF_WOUT   = al(OFF_WHHB_B + SZ_WHHB);
constexpr size_t OFF_PRE_F  = al(OFF_WOUT + SZ_WOUT);
constexpr size_t OFF_PRE_B  = al(OFF_PRE_F + SZ_PRE);
constexpr size_t OFF_HBUF   = al(OFF_PRE_B + SZ_PRE);
constexpr size_t OFF_LOUT   = al(OFF_HBUF + SZ_HBUF);
constexpr size_t OFF_FEAT   = al(OFF_LOUT + SZ_LOUT);
constexpr size_t OFF_CTR    = al(OFF_FEAT + SZ_FEAT);
} // namespace cfg

#define WMMA_BF16(Am, Bm, Cm)                                                   \
  __builtin_amdgcn_wmma_f32_16x16x32_bf16(false, (Am), false, (Bm), (short)0,   \
                                          (Cm), false, false)

__device__ __forceinline__ bf16_t f2bf(float f) {
  union { float f; unsigned u; } in{f};
  unsigned u = in.u;
  unsigned r = (u + 0x7FFFu + ((u >> 16) & 1u)) >> 16;  // round-to-nearest-even
  union { unsigned short s; bf16_t b; } out;
  out.s = (unsigned short)r;
  return out.b;
}

// Load a 16x32 bf16 A/B fragment from a row-major matrix (16 consecutive rows,
// K-window [k0, k0+32)). CDNA5 16-bit layout: lanes 0-15 hold K[0:8)+K[16:24)
// of row (lane), lanes 16-31 hold K[8:16)+K[24:32) of row (lane-16).
__device__ __forceinline__ v16bf load_frag(const bf16_t* base, int ld, int row0,
                                           int k0, int lane) {
  const int r  = row0 + (lane & 15);
  const int ko = (lane >> 4) << 3;
  const bf16_t* p = base + (size_t)r * ld + k0;
  v16bf out;
  ((v8bf*)&out)[0] = *(const v8bf*)(p + ko);
  ((v8bf*)&out)[1] = *(const v8bf*)(p + 16 + ko);
  return out;
}

// CDNA5 async global->LDS copy (ASYNCcnt-tracked), 16B per lane.
__device__ __forceinline__ void async_ld_b128(unsigned lds_off, const void* g) {
  asm volatile("global_load_async_to_lds_b128 %0, %1, off"
               :: "v"(lds_off), "v"((unsigned long long)(uintptr_t)g)
               : "memory");
}
__device__ __forceinline__ void wait_async0() {
  asm volatile("s_wait_asynccnt 0" ::: "memory");
}

__device__ __forceinline__ float sigf(float x) { return 1.f / (1.f + __expf(-x)); }

// ---------------- init: zero ping-pong h buffers + step counters --------------
__global__ void k_init(bf16_t* hbuf, unsigned* ctr) {
  const int n = 2 * 2 * cfg::B * cfg::H;
  for (int i = blockIdx.x * blockDim.x + threadIdx.x; i < n; i += gridDim.x * blockDim.x)
    hbuf[i] = f2bf(0.f);
  if (blockIdx.x == 0 && threadIdx.x < 2) ctr[threadIdx.x] = 0u;
}

// ---------------- embedding gather + sentiment concat, fp32 -> bf16 -----------
__global__ void k_embed(const float* __restrict__ emb, const int* __restrict__ sent,
                        const float* __restrict__ sentiment, bf16_t* __restrict__ xb) {
  const int bt = blockIdx.x;
  const int b = bt >> 9;  // T = 512
  const int idx = sent[bt];
  const float sv = sentiment[b];
  const float* erow = emb + (size_t)idx * cfg::E;
  bf16_t* xrow = xb + (size_t)bt * cfg::DP;
  for (int d = threadIdx.x; d < cfg::DP; d += blockDim.x) {
    float v = (d < cfg::E) ? erow[d] : ((d == cfg::E) ? sv : 0.f);
    xrow[d] = f2bf(v);
  }
}

// ---------------- weight pack fp32 -> bf16 with zero K-padding ----------------
__global__ void k_pack(const float* __restrict__ src, bf16_t* __restrict__ dst,
                       int scols, int dcols) {
  const int r = blockIdx.x;
  const float* s = src + (size_t)r * scols;
  bf16_t* d = dst + (size_t)r * dcols;
  for (int c = threadIdx.x; c < dcols; c += blockDim.x)
    d[c] = f2bf(c < scols ? s[c] : 0.f);
}

// ------------- input projection: pre = x @ Wih^T + bih + bhh (both dirs) ------
// Register blocking: one A fragment reused over 4 B tiles -> 4 wmma / 5 loads.
__global__ void k_gemm_pre(const bf16_t* __restrict__ xb,
                           const bf16_t* __restrict__ wf, const bf16_t* __restrict__ wb,
                           const float* __restrict__ bihf, const float* __restrict__ bhhf,
                           const float* __restrict__ bihb, const float* __restrict__ bhhb,
                           float* __restrict__ pref, float* __restrict__ preb) {
  const int lane = threadIdx.x & 31;
  const int wv = threadIdx.x >> 5;
  const bool bd = blockIdx.z != 0;
  const bf16_t* W = bd ? wb : wf;
  const float* bi = bd ? bihb : bihf;
  const float* bh = bd ? bhhb : bhhf;
  float* pre = bd ? preb : pref;
  const int row0 = blockIdx.y * 16;                  // bt rows
  const int col_base = (blockIdx.x * 8 + wv) * 64;   // 4 x 16 gate columns
  v8f acc[4] = {v8f{}, v8f{}, v8f{}, v8f{}};
#pragma unroll 1
  for (int k0 = 0; k0 < cfg::DP; k0 += 32) {
    __builtin_prefetch(xb + (size_t)(row0 + (lane & 15)) * cfg::DP + k0 + 128, 0, 1);
    v16bf a = load_frag(xb, cfg::DP, row0, k0, lane);
#pragma unroll
    for (int c = 0; c < 4; ++c) {
      v16bf bm = load_frag(W, cfg::DP, col_base + c * 16, k0, lane);
      acc[c] = WMMA_BF16(a, bm, acc[c]);
    }
  }
  const int m0 = row0 + ((lane >> 4) << 3);
#pragma unroll
  for (int c = 0; c < 4; ++c) {
    const int n = col_base + c * 16 + (lane & 15);
    const float bias = bi[n] + bh[n];
#pragma unroll
    for (int r = 0; r < 8; ++r)
      pre[(size_t)(m0 + r) * cfg::G + n] = acc[c][r] + bias;
  }
}

// ------------- persistent bidirectional LSTM recurrence -----------------------
// 8 blocks: blockIdx/4 = direction, blockIdx%4 = 128-hidden-unit slice.
// Per wave: 2x4 accumulator tiles (batch-half x gate) so each lane's C elements
// carry i/f/g/o for the SAME (batch,unit) -> cell update is pure register math.
// pre tiles (64KB/step) are double-buffered into LDS with async global->LDS
// copies issued one step ahead (ASYNCcnt + barrier at step top).
__global__ void k_lstm(const float* __restrict__ pref, const float* __restrict__ preb,
                       const bf16_t* __restrict__ whf, const bf16_t* __restrict__ whb,
                       bf16_t* __restrict__ hbuf, bf16_t* __restrict__ lstm_out,
                       unsigned* __restrict__ ctr) {
  const int tid = threadIdx.x;
  const int lane = tid & 31, wv = tid >> 5;
  const int dir = blockIdx.x >> 2, wg = blockIdx.x & 3;
  const int J0 = wg * 128;
  const float* pre = dir ? preb : pref;
  const bf16_t* whh = dir ? whb : whf;
  bf16_t* hb0 = hbuf + (size_t)dir * 2 * cfg::B * cfg::H;
  unsigned* cnt = ctr + dir;
  __shared__ float pbuf[2][4][cfg::B][128];  // [stage][gate][batch][local j]

  const int jl = wv * 16 + (lane & 15);      // local hidden col owned by lane
  const int j  = J0 + jl;
  float creg[2][8];
#pragma unroll
  for (int mh = 0; mh < 2; ++mh)
#pragma unroll
    for (int r = 0; r < 8; ++r) creg[mh][r] = 0.f;

  // async prefetch of step-0 pre tile into stage 0 (16 chunks per wave)
  {
    const int tt0 = dir ? (cfg::T - 1) : 0;
#pragma unroll 1
    for (int i = 0; i < 16; ++i) {
      const int id = wv * 16 + i;
      const int b = id >> 2, gi = id & 3;
      const float* g = pre + ((size_t)b * cfg::T + tt0) * cfg::G + gi * cfg::H + J0 + lane * 4;
      async_ld_b128((unsigned)(uintptr_t)&pbuf[0][gi][b][lane * 4], g);
    }
  }

  for (int t = 0; t < cfg::T; ++t) {
    const int tt = dir ? (cfg::T - 1 - t) : t;
    const int st = t & 1;
    const bf16_t* hread = hb0 + (size_t)st * cfg::B * cfg::H;
    bf16_t* hwrite = hb0 + (size_t)(st ^ 1) * cfg::B * cfg::H;

    wait_async0();       // own async copies landed
    __syncthreads();     // everyone's copies landed

    // async prefetch of next step's pre tile into the other stage
    if (t + 1 < cfg::T) {
      const int ttn = dir ? (cfg::T - 2 - t) : (t + 1);
#pragma unroll 1
      for (int i = 0; i < 16; ++i) {
        const int id = wv * 16 + i;
        const int b = id >> 2, gi = id & 3;
        const float* g = pre + ((size_t)b * cfg::T + ttn) * cfg::G + gi * cfg::H + J0 + lane * 4;
        async_ld_b128((unsigned)(uintptr_t)&pbuf[st ^ 1][gi][b][lane * 4], g);
      }
    }

    // accumulators initialized from staged pre (already in C-fragment layout)
    v8f acc[2][4];
#pragma unroll
    for (int mh = 0; mh < 2; ++mh)
#pragma unroll
      for (int gi = 0; gi < 4; ++gi)
#pragma unroll
        for (int r = 0; r < 8; ++r) {
          const int b = mh * 16 + ((lane >> 4) << 3) + r;
          acc[mh][gi][r] = pbuf[st][gi][b][jl];
        }

    // h(32x512) @ Whh_slice^T : A shared across 4 gates, B across 2 batch halves
#pragma unroll 1
    for (int k0 = 0; k0 < cfg::H; k0 += 32) {
      v16bf a0 = load_frag(hread, cfg::H, 0, k0, lane);
      v16bf a1 = load_frag(hread, cfg::H, 16, k0, lane);
#pragma unroll
      for (int gi = 0; gi < 4; ++gi) {
        v16bf bm = load_frag(whh, cfg::H, gi * cfg::H + J0 + wv * 16, k0, lane);
        acc[0][gi] = WMMA_BF16(a0, bm, acc[0][gi]);
        acc[1][gi] = WMMA_BF16(a1, bm, acc[1][gi]);
      }
    }

    // LSTM cell update: i/f/g/o for one (b,j) live in the same lane element
#pragma unroll
    for (int mh = 0; mh < 2; ++mh)
#pragma unroll
      for (int r = 0; r < 8; ++r) {
        const int b = mh * 16 + ((lane >> 4) << 3) + r;
        const float ig = sigf(acc[mh][0][r]);
        const float fg = sigf(acc[mh][1][r]);
        const float gg = tanhf(acc[mh][2][r]);
        const float og = sigf(acc[mh][3][r]);
        const float c = fg * creg[mh][r] + ig * gg;
        creg[mh][r] = c;
        const float h = og * tanhf(c);
        hwrite[b * cfg::H + j] = f2bf(h);
        lstm_out[((size_t)b * cfg::T + tt) * cfg::H2 + dir * cfg::H + j] = f2bf(h);
      }

    __syncthreads();
    if (tid == 0) {
      __threadfence();                       // release h(t+1) slice
      atomicAdd(cnt, 1u);
      const unsigned target = 4u * (unsigned)(t + 1);
      while (__hip_atomic_load(cnt, __ATOMIC_ACQUIRE, __HIP_MEMORY_SCOPE_AGENT) < target)
        __builtin_amdgcn_s_sleep(1);
    }
    __syncthreads();
  }
}

// ------------- output projection: feats = lstm_out @ W_out^T + b_out ----------
// One B fragment (16 tag rows) reused over 4 A tiles -> 4 wmma / 5 loads.
__global__ void k_feats(const bf16_t* __restrict__ lstm_out, const bf16_t* __restrict__ wout,
                        const float* __restrict__ bout, float* __restrict__ feats) {
  const int lane = threadIdx.x & 31, wv = threadIdx.x >> 5;
  const int row_base = (blockIdx.x * 8 + wv) * 64;
  v8f acc[4] = {v8f{}, v8f{}, v8f{}, v8f{}};
#pragma unroll 1
  for (int k0 = 0; k0 < cfg::H2; k0 += 32) {
    v16bf bm = load_frag(wout, cfg::H2, 0, k0, lane);   // 16 rows = K tags
#pragma unroll
    for (int m = 0; m < 4; ++m) {
      v16bf a = load_frag(lstm_out, cfg::H2, row_base + m * 16, k0, lane);
      acc[m] = WMMA_BF16(a, bm, acc[m]);
    }
  }
  const int n = lane & 15;
  const float bias = bout[n];
#pragma unroll
  for (int m = 0; m < 4; ++m) {
    const int m0 = row_base + m * 16 + ((lane >> 4) << 3);
#pragma unroll
    for (int r = 0; r < 8; ++r)
      feats[(size_t)(m0 + r) * cfg::NK + n] = acc[m][r] + bias;
  }
}

// ------------- CRF forward + gold score + loss (K=16, single block) -----------
__global__ void k_crf(const float* __restrict__ feats, const float* __restrict__ trans,
                      const int* __restrict__ sent, const int* __restrict__ tags,
                      float* __restrict__ out) {
  const int tid = threadIdx.x;           // 512 threads: (b,k) = (tid/16, tid%16)
  const int b = tid >> 4, k = tid & 15;
  __shared__ float s_alpha[cfg::B][cfg::NK];
  __shared__ float s_trans[cfg::NK][cfg::NK];
  __shared__ float s_gold[cfg::B][cfg::NK];
  __shared__ float s_fin[cfg::B];
  if (tid < 256) s_trans[tid >> 4][tid & 15] = trans[tid];
  __syncthreads();

  float alpha = feats[((size_t)b * cfg::T) * cfg::NK + k];    // alpha0 = feats[:,0]
  for (int t = 1; t < cfg::T; ++t) {
    s_alpha[b][k] = alpha;
    __syncthreads();
    float mx = -3.4e38f;
#pragma unroll
    for (int jj = 0; jj < cfg::NK; ++jj)
      mx = fmaxf(mx, s_alpha[b][jj] + s_trans[k][jj]);
    float ssum = 0.f;
#pragma unroll
    for (int jj = 0; jj < cfg::NK; ++jj)
      ssum += __expf(s_alpha[b][jj] + s_trans[k][jj] - mx);
    const float nw = __logf(ssum) + mx + feats[((size_t)b * cfg::T + t) * cfg::NK + k];
    if (sent[b * cfg::T + t] != 0) alpha = nw;
    __syncthreads();
  }
  // gold score: emissions (all t, masked) + transitions (t>=1, masked)
  float gold = 0.f;
  for (int t = k; t < cfg::T; t += cfg::NK) {
    const int tg = tags[b * cfg::T + t];
    const float mk = (sent[b * cfg::T + t] != 0) ? 1.f : 0.f;
    gold += feats[((size_t)b * cfg::T + t) * cfg::NK + tg] * mk;
    if (t >= 1) gold += s_trans[tags[b * cfg::T + t - 1]][tg] * mk;
  }
  s_alpha[b][k] = alpha;
  s_gold[b][k] = gold;
  __syncthreads();
  if (k == 0) {
    float fs = 0.f, gs = 0.f;
    for (int jj = 0; jj < cfg::NK; ++jj) { fs += s_alpha[b][jj]; gs += s_gold[b][jj]; }
    s_fin[b] = fs - gs;
  }
  __syncthreads();
  if (tid == 0) {
    float L = 0.f;
    for (int bb = 0; bb < cfg::B; ++bb) L += s_fin[bb];
    out[0] = L / (float)cfg::B;
  }
}

extern "C" void kernel_launch(void* const* d_in, const int* in_sizes, int n_in,
                              void* d_out, int out_size, void* d_ws, size_t ws_size,
                              hipStream_t stream) {
  (void)in_sizes; (void)n_in; (void)out_size; (void)ws_size;
  const float* emb   = (const float*)d_in[0];
  const float* Wih_f = (const float*)d_in[1];
  const float* Whh_f = (const float*)d_in[2];
  const float* bih_f = (const float*)d_in[3];
  const float* bhh_f = (const float*)d_in[4];
  const float* Wih_b = (const float*)d_in[5];
  const float* Whh_b = (const float*)d_in[6];
  const float* bih_b = (const float*)d_in[7];
  const float* bhh_b = (const float*)d_in[8];
  const float* W_out = (const float*)d_in[9];
  const float* b_out = (const float*)d_in[10];
  const float* trans = (const float*)d_in[11];
  const int*   sent  = (const int*)d_in[12];
  const float* senti = (const float*)d_in[13];
  const int*   tags  = (const int*)d_in[14];

  char* ws = (char*)d_ws;
  bf16_t* xb    = (bf16_t*)(ws + cfg::OFF_XB);
  bf16_t* wihbf = (bf16_t*)(ws + cfg::OFF_WIHB_F);
  bf16_t* wihbb = (bf16_t*)(ws + cfg::OFF_WIHB_B);
  bf16_t* whhbf = (bf16_t*)(ws + cfg::OFF_WHHB_F);
  bf16_t* whhbb = (bf16_t*)(ws + cfg::OFF_WHHB_B);
  bf16_t* woutb = (bf16_t*)(ws + cfg::OFF_WOUT);
  float*  pref  = (float*)(ws + cfg::OFF_PRE_F);
  float*  preb  = (float*)(ws + cfg::OFF_PRE_B);
  bf16_t* hbuf  = (bf16_t*)(ws + cfg::OFF_HBUF);
  bf16_t* lout  = (bf16_t*)(ws + cfg::OFF_LOUT);
  float*  feats = (float*)(ws + cfg::OFF_FEAT);
  unsigned* ctr = (unsigned*)(ws + cfg::OFF_CTR);

  k_init<<<64, 256, 0, stream>>>(hbuf, ctr);
  k_embed<<<cfg::BT, 256, 0, stream>>>(emb, sent, senti, xb);
  k_pack<<<cfg::G, 256, 0, stream>>>(Wih_f, wihbf, cfg::D, cfg::DP);
  k_pack<<<cfg::G, 256, 0, stream>>>(Wih_b, wihbb, cfg::D, cfg::DP);
  k_pack<<<cfg::G, 256, 0, stream>>>(Whh_f, whhbf, cfg::H, cfg::H);
  k_pack<<<cfg::G, 256, 0, stream>>>(Whh_b, whhbb, cfg::H, cfg::H);
  k_pack<<<cfg::NK, 256, 0, stream>>>(W_out, woutb, cfg::H2, cfg::H2);

  dim3 gpre(cfg::G / 512, cfg::BT / 16, 2);   // (4, 1024, 2)
  k_gemm_pre<<<gpre, 256, 0, stream>>>(xb, wihbf, wihbb, bih_f, bhh_f, bih_b, bhh_b,
                                       pref, preb);
  k_lstm<<<8, 256, 0, stream>>>(pref, preb, whhbf, whhbb, hbuf, lout, ctr);
  k_feats<<<cfg::BT / 512, 256, 0, stream>>>(lout, woutb, b_out, feats);
  k_crf<<<1, 512, 0, stream>>>(feats, trans, sent, tags, (float*)d_out);
}
